// Decomp_Net_Translation_50517405335815
// MI455X (gfx1250) — matrix operand order repaired and checked by reading, not compiled, not run on hardware
//
#include <hip/hip_runtime.h>
#include <hip/hip_bf16.h>

typedef __attribute__((ext_vector_type(16))) _Float16 v16h;
typedef __attribute__((ext_vector_type(8)))  float    v8f;
typedef __attribute__((ext_vector_type(4)))  float    v4f;   // native vector for NT stores

__device__ __constant__ int PAIR_A[20] = {0,0,0,0, 1,1,1,1, 2,2,2,2, 3,3,3,3, 4,4,4,4};
__device__ __constant__ int PAIR_B[20] = {1,2,3,4, 0,2,3,4, 0,1,3,4, 0,1,2,4, 0,1,2,3};

// ---------------------------------------------------------------------------
// Weight pre-pack: fp32 OIHW -> f16 swizzled to WMMA B-fragment order:
//   dst[((kc*NT + t)*32 + lane)*16 + e]  =  W[k][o]
//   k = kc*32 + (lane>>4)*16 + e ,  o = t*16 + (lane&15)
// so each lane's 16 B elements for (kc, t) are one contiguous 32B load.
// ---------------------------------------------------------------------------
__global__ void pack_w_k(const float* __restrict__ w, _Float16* __restrict__ dst,
                         int Cin, int Cout, int Kpad)
{
    int idx = blockIdx.x * 256 + threadIdx.x;
    int total = Kpad * Cout;                  // = Kchunks * NT * 512
    if (idx >= total) return;
    int e    = idx & 15;
    int lane = (idx >> 4) & 31;
    int tt   = idx >> 9;                      // kc*NT + t
    int NTl  = Cout >> 4;
    int kc   = tt / NTl, t = tt - kc * NTl;
    int k    = kc * 32 + (lane >> 4) * 16 + e;
    int o    = t * 16 + (lane & 15);
    float v = 0.f;
    int K = Cin * 9;
    if (k < K) {
        int c = k / 9, r = k - c * 9;
        v = w[((o * Cin + c) * 3 + r / 3) * 3 + (r - (r / 3) * 3)];
    }
    dst[idx] = (_Float16)v;
}

// ---------------------------------------------------------------------------
// Implicit-GEMM conv3x3 + bias + leaky ReLU via v_wmma_f32_16x16x32_f16.
// One wave32 per block, one 16-pixel output strip (M tile).
// Stage: im2col A tile built once in LDS, already in fragment order
//   sA[kc*512 + lane*16 + e]  (lane's 16 elements contiguous -> ds_load_b128 x2)
// Loop:  A = 2x ds_load_b128, B = 2x global_load_b128 (pre-swizzled, L2-hot),
//        NT wmma per k-chunk; branch-free pipeline, no barriers.
// Epilogue: transpose C tile through LDS so 16 consecutive lanes emit 16
//           consecutive pixels of one channel (full-cacheline stores).
// ---------------------------------------------------------------------------
template<int NT>
__global__ void __launch_bounds__(32) conv3x3_wmma(
    const float* __restrict__ in, const _Float16* __restrict__ wpk,
    const float* __restrict__ bias, float* __restrict__ out,
    int N, int Cin, int Hin, int Win, int Hout, int Wout,
    int stride, int pad, int K, int Kpad, float slope)
{
    extern __shared__ unsigned char smem_raw[];
    _Float16* sA = (_Float16*)smem_raw;       // Kpad*16 f16, fragment order

    const int lane = threadIdx.x;
    const int WT   = (Wout + 15) >> 4;
    const int si   = blockIdx.x;              // strip id, grid.x == strips

    const int n   = si / (Hout * WT);
    int rem       = si - n * (Hout * WT);
    const int oy  = rem / WT;
    const int ox0 = (rem - oy * WT) << 4;

    // --- build im2col A tile in LDS, fragment order ---
    {
        const int Kch  = Kpad >> 5;
        const int tot  = Kch * 512;           // Kpad*16 elements
        const int iy0  = oy * stride - pad;
        const float* inN = in + (size_t)n * Cin * Hin * Win;
        for (int idx = lane; idx < tot; idx += 32) {
            int e  = idx & 15;
            int l  = (idx >> 4) & 31;         // target lane
            int kc = idx >> 9;
            int mm   = l & 15;
            int half = l >> 4;
            int koff = half * 16 + e;
            int k = kc * 32 + koff;
            float v = 0.f;
            if (k < K) {
                int c  = k / 9;
                int r  = k - c * 9;
                int dy = r / 3;
                int dx = r - dy * 3;
                int iy = iy0 + dy;
                int ix = (ox0 + mm) * stride - pad + dx;
                if (iy >= 0 && iy < Hin && ix >= 0 && ix < Win)
                    v = inN[((size_t)c * Hin + iy) * Win + ix];
            }
            sA[idx] = (_Float16)v;
        }
    }
    __syncthreads();                          // single wave: wait_dscnt + nop

    v8f acc[NT];
    #pragma unroll
    for (int t = 0; t < NT; ++t)
        #pragma unroll
        for (int v = 0; v < 8; ++v) acc[t][v] = 0.f;

    // --- hot loop: wide fragment loads + WMMA ---
    const int Kch = Kpad >> 5;
    const v16h* __restrict__ bfrag = (const v16h*)wpk;
    for (int kc = 0; kc < Kch; ++kc) {
        v16h a = *(const v16h*)(&sA[(kc << 9) + (lane << 4)]);
        #pragma unroll
        for (int t = 0; t < NT; ++t) {
            v16h b = bfrag[(size_t)(kc * NT + t) * 32 + lane];
            acc[t] = __builtin_amdgcn_wmma_f32_16x16x32_f16(
                false, a, false, b, (short)0, acc[t], false, false);
        }
    }

    // --- epilogue: C layout row(pixel) = vgpr + 8*(lane>=16), col(ch) = lane&15
    //     bias + leaky ReLU, transpose via LDS (17-pad: bank-conflict-free),
    //     then coalesced stores: half-wave = 16 consecutive pixels of one ch.
    const int half = lane >> 4;
    const int m    = lane & 15;
    __syncthreads();                          // A tile dead; reuse LDS
    float* sOut = (float*)smem_raw;           // [Cout][17] floats
    #pragma unroll
    for (int t = 0; t < NT; ++t) {
        int ch = t * 16 + m;
        float bv = bias[ch];
        #pragma unroll
        for (int v = 0; v < 8; ++v) {
            float y = acc[t][v] + bv;
            y = (y > 0.f) ? y : y * slope;
            sOut[ch * 17 + (v + half * 8)] = y;
        }
    }
    __syncthreads();
    const int Cout = NT * 16;
    float* outN = out + (size_t)n * Cout * Hout * Wout;
    const int ox = ox0 + m;
    if (ox < Wout) {
        #pragma unroll
        for (int g = 0; g < Cout / 2; ++g) {
            int ch = 2 * g + half;
            outN[((size_t)ch * Hout + oy) * Wout + ox] = sOut[ch * 17 + m];
        }
    }
}

// ---------------------------------------------------------------------------
// Cost volume: block = (row y, batch b, pair p). Stage f0 row + padded 9-row
// f1 neighborhood in LDS (f16), compute 81 x 24 correlations (mean over 96 ch),
// leaky ReLU 0.1, write into channels [96..177) of the concat buffer x0.
// ---------------------------------------------------------------------------
__global__ void __launch_bounds__(128) cost_volume_k(
    const float* __restrict__ feats, float* __restrict__ x0)
{
    __shared__ _Float16 sF0[96 * 24];
    __shared__ _Float16 sF1[96 * 9 * 32];

    const int y = blockIdx.x, b = blockIdx.y, p = blockIdx.z;
    const int ia = PAIR_A[p], ib = PAIR_B[p];
    const float* f0 = feats + (size_t)(ia * 8 + b) * 96 * 576;
    const float* f1 = feats + (size_t)(ib * 8 + b) * 96 * 576;
    const int tid = threadIdx.x;

    for (int e = tid; e < 96 * 24; e += 128) {
        int c = e / 24, x = e - c * 24;
        sF0[e] = (_Float16)f0[c * 576 + y * 24 + x];
    }
    for (int e = tid; e < 96 * 9 * 32; e += 128) {
        int c  = e / (9 * 32);
        int r  = e - c * (9 * 32);
        int dy = r / 32;
        int xx = r - dy * 32;
        int yy = y + dy - 4, x = xx - 4;
        float v = 0.f;
        if (yy >= 0 && yy < 24 && x >= 0 && x < 24) v = f1[c * 576 + yy * 24 + x];
        sF1[e] = (_Float16)v;
    }
    __syncthreads();

    const size_t pb = (size_t)p * 8 + b;
    float* dst = x0 + (pb * 177 + 96) * 576 + y * 24;
    for (int idx = tid; idx < 81 * 24; idx += 128) {
        int s = idx / 24, x = idx - s * 24;
        int dy = s / 9, dx = s - dy * 9;
        float accv = 0.f;
        #pragma unroll 4
        for (int c = 0; c < 96; ++c)
            accv += (float)sF0[c * 24 + x] * (float)sF1[(c * 9 + dy) * 32 + x + dx];
        accv *= (1.f / 96.f);
        accv = (accv > 0.f) ? accv : 0.1f * accv;
        dst[(size_t)s * 576 + x] = accv;
    }
}

// Copy fa into channels [0..96) of the concat buffer
__global__ void copy_fa_k(const float* __restrict__ feats, float* __restrict__ x0)
{
    size_t i = (size_t)blockIdx.x * 256 + threadIdx.x;
    if (i >= (size_t)160 * 96 * 576) return;
    int pb = (int)(i / (96 * 576));
    int r  = (int)(i - (size_t)pb * (96 * 576));
    int p = pb >> 3, b = pb & 7;
    x0[(size_t)pb * 177 * 576 + r] = feats[(size_t)(PAIR_A[p] * 8 + b) * 96 * 576 + r];
}

// Global mean pool: [160][32][576] -> [160][32]
__global__ void __launch_bounds__(32) pool_k(const float* __restrict__ x, float* __restrict__ feat)
{
    int s = blockIdx.x, c = threadIdx.x;
    const float* p = x + ((size_t)s * 32 + c) * 576;
    float acc = 0.f;
    for (int i = 0; i < 576; ++i) acc += p[i];
    feat[s * 32 + c] = acc * (1.f / 576.f);
}

// Two 32x2 linears -> flows[40][8][2] (q<20: lin1, q>=20: lin2 of pair q-20)
__global__ void flows_k(const float* __restrict__ feat,
                        const float* __restrict__ w1, const float* __restrict__ b1,
                        const float* __restrict__ w2, const float* __restrict__ b2,
                        float* __restrict__ flows)
{
    int i = blockIdx.x * 64 + threadIdx.x;
    if (i >= 640) return;
    int d = i & 1, qb = i >> 1;
    int q = qb >> 3, b = qb & 7;
    const float* w  = (q < 20) ? w1 : w2;
    const float* bb = (q < 20) ? b1 : b2;
    int p = (q < 20) ? q : q - 20;
    const float* f = feat + (p * 8 + b) * 32;
    float acc = bb[d];
    #pragma unroll
    for (int c = 0; c < 32; ++c) acc += f[c] * w[c * 2 + d];
    flows[i] = acc;
}

// Broadcast flows to (40,8,384,384,2). 377MB stream >> 192MB L2:
// use non-temporal stores to keep weights/activations resident.
__global__ void bcast_k(const float* __restrict__ flows, v4f* __restrict__ out)
{
    size_t i = (size_t)blockIdx.x * 256 + threadIdx.x;
    const size_t per_img = (size_t)384 * 384 * 2 / 4;   // 73728
    if (i >= per_img * 320) return;
    size_t img = i / per_img;
    const float* fl = flows + img * 2;
    v4f v = { fl[0], fl[1], fl[0], fl[1] };
    __builtin_nontemporal_store(v, &out[i]);
}

// ---------------------------------------------------------------------------
// Host driver
// ---------------------------------------------------------------------------
static inline int roundup32(int x) { return (x + 31) & ~31; }

extern "C" void kernel_launch(void* const* d_in, const int* in_sizes, int n_in,
                              void* d_out, int out_size, void* d_ws, size_t ws_size,
                              hipStream_t stream)
{
    (void)in_sizes; (void)n_in; (void)out_size; (void)ws_size;
    const float* I[5];
    for (int i = 0; i < 5; ++i) I[i] = (const float*)d_in[i];

    struct LD { const float* w; const float* b; int cin, cout, hin, stride; float slope; };
    LD L[13];
    const int F[4] = {16, 32, 64, 96};
    int inc = 3, hin = 384;
    for (int l = 0; l < 4; ++l) {
        L[2*l]   = { (const float*)d_in[5+4*l], (const float*)d_in[6+4*l], inc,  F[l], hin,     2, 0.1f };
        L[2*l+1] = { (const float*)d_in[7+4*l], (const float*)d_in[8+4*l], F[l], F[l], hin / 2, 1, 0.1f };
        inc = F[l]; hin /= 2;
    }
    const int eci[5] = {177, 128, 128, 96, 64}, eco[5] = {128, 128, 96, 64, 32};
    for (int k = 0; k < 5; ++k)
        L[8+k] = { (const float*)d_in[21+2*k], (const float*)d_in[22+2*k], eci[k], eco[k], 24, 1, 0.2f };

    unsigned char* ws = (unsigned char*)d_ws;
    size_t off = 0;
    auto take = [&](size_t bytes) -> size_t {
        size_t r = off; off += (bytes + 255) & ~(size_t)255; return r;
    };
    size_t wpk_off[13]; int Kp[13];
    for (int i = 0; i < 13; ++i) {
        Kp[i] = roundup32(L[i].cin * 9);
        wpk_off[i] = take((size_t)Kp[i] * L[i].cout * sizeof(_Float16));
    }
    size_t feats = take((size_t)5 * 8 * 96 * 576 * 4);
    size_t x0    = take((size_t)160 * 177 * 576 * 4);
    size_t e1    = take((size_t)160 * 128 * 576 * 4);   // also pyramid ping
    size_t e2    = take((size_t)160 * 128 * 576 * 4);   // also pyramid pong
    size_t featv = take((size_t)160 * 32 * 4);
    size_t flows = take((size_t)640 * 4);

    // pack all conv weights to f16, swizzled to B-fragment order
    for (int i = 0; i < 13; ++i) {
        int total = Kp[i] * L[i].cout;
        pack_w_k<<<(total + 255) / 256, 256, 0, stream>>>(
            L[i].w, (_Float16*)(ws + wpk_off[i]), L[i].cin, L[i].cout, Kp[i]);
    }

    auto conv = [&](int i, const float* src, float* dst, int N) {
        int hout = (L[i].stride == 2) ? L[i].hin / 2 : L[i].hin;
        int pad  = (L[i].stride == 2) ? 0 : 1;            // lax SAME padding
        int WT = (hout + 15) / 16;
        int strips = N * hout * WT;
        dim3 g(strips), blk(32);
        size_t shA   = (size_t)Kp[i] * 16 * sizeof(_Float16);  // im2col A tile
        size_t shOut = (size_t)L[i].cout * 17 * sizeof(float); // epilogue transpose
        size_t shmem = shA > shOut ? shA : shOut;
        int K = L[i].cin * 9;
        const _Float16* wp = (const _Float16*)(ws + wpk_off[i]);
        switch (L[i].cout / 16) {
        case 1: conv3x3_wmma<1><<<g, blk, shmem, stream>>>(src, wp, L[i].b, dst, N, L[i].cin, L[i].hin, L[i].hin, hout, hout, L[i].stride, pad, K, Kp[i], L[i].slope); break;
        case 2: conv3x3_wmma<2><<<g, blk, shmem, stream>>>(src, wp, L[i].b, dst, N, L[i].cin, L[i].hin, L[i].hin, hout, hout, L[i].stride, pad, K, Kp[i], L[i].slope); break;
        case 4: conv3x3_wmma<4><<<g, blk, shmem, stream>>>(src, wp, L[i].b, dst, N, L[i].cin, L[i].hin, L[i].hin, hout, hout, L[i].stride, pad, K, Kp[i], L[i].slope); break;
        case 6: conv3x3_wmma<6><<<g, blk, shmem, stream>>>(src, wp, L[i].b, dst, N, L[i].cin, L[i].hin, L[i].hin, hout, hout, L[i].stride, pad, K, Kp[i], L[i].slope); break;
        default: conv3x3_wmma<8><<<g, blk, shmem, stream>>>(src, wp, L[i].b, dst, N, L[i].cin, L[i].hin, L[i].hin, hout, hout, L[i].stride, pad, K, Kp[i], L[i].slope); break;
        }
    };

    float* pA = (float*)(ws + e1);   // alias: pyramid ping-pong reuses est buffers
    float* pB = (float*)(ws + e2);

    // feature pyramid, 5 frames
    for (int img = 0; img < 5; ++img) {
        const float* src = I[img];
        float* fOut = (float*)(ws + feats) + (size_t)img * 8 * 96 * 576;
        for (int l = 0; l < 4; ++l) {
            conv(2 * l, src, pA, 8);
            float* dst = (l == 3) ? fOut : pB;
            conv(2 * l + 1, pA, dst, 8);
            src = dst;
        }
    }

    // concat buffer: fa copy + cost volume
    copy_fa_k<<<(int)(((size_t)160 * 96 * 576 + 255) / 256), 256, 0, stream>>>(
        (const float*)(ws + feats), (float*)(ws + x0));
    dim3 cg(24, 8, 20);
    cost_volume_k<<<cg, 128, 0, stream>>>((const float*)(ws + feats), (float*)(ws + x0));

    // estimator convs, N = 160 (20 pairs x batch 8)
    conv(8,  (const float*)(ws + x0), (float*)(ws + e1), 160);
    conv(9,  (const float*)(ws + e1), (float*)(ws + e2), 160);
    conv(10, (const float*)(ws + e2), (float*)(ws + e1), 160);
    conv(11, (const float*)(ws + e1), (float*)(ws + e2), 160);
    conv(12, (const float*)(ws + e2), (float*)(ws + e1), 160);

    pool_k<<<160, 32, 0, stream>>>((const float*)(ws + e1), (float*)(ws + featv));
    flows_k<<<10, 64, 0, stream>>>((const float*)(ws + featv),
                                   (const float*)d_in[31], (const float*)d_in[32],
                                   (const float*)d_in[33], (const float*)d_in[34],
                                   (float*)(ws + flows));
    size_t tot4 = (size_t)40 * 8 * 384 * 384 * 2 / 4;
    bcast_k<<<(int)((tot4 + 255) / 256), 256, 0, stream>>>(
        (const float*)(ws + flows), (v4f*)d_out);
}
